// Synthesis_17755394801865
// MI455X (gfx1250) — matrix operand order repaired and checked by reading, not compiled, not run on hardware
//
#include <hip/hip_runtime.h>
#include <hip/hip_bf16.h>
#include <math.h>
#include <stdint.h>

typedef _Float16 half_t;
typedef __attribute__((ext_vector_type(16))) _Float16 v16h;
typedef __attribute__((ext_vector_type(8)))  _Float16 v8h;
typedef __attribute__((ext_vector_type(8)))  float    v8f;
typedef __attribute__((ext_vector_type(4)))  unsigned int u32x4;
typedef __attribute__((ext_vector_type(8)))  int i32x8;
typedef __attribute__((ext_vector_type(4)))  int i32x4;

#define B_    8
#define CIN   512
#define COUT  256
#define HW    64
#define OHW   128
#define KC_N  16          // 512 / 32 K-chunks

#define ROW_H  (66 * 32)           // padded LDS row: 66 cols x 32 ch (halves)
#define BUF_H  (3 * ROW_H)         // one LDS buffer (halves)
#define ROW_B  (ROW_H * 2)         // bytes
#define BUF_B  (BUF_H * 2)         // bytes

// ---------------------------------------------------------------------------
// Prep kernel 1: s[b,i] = style[b,:] . mod_w[i,:] / sqrt(512) + mod_b[i]
// ---------------------------------------------------------------------------
__global__ void mod_kernel(const float* __restrict__ style,
                           const float* __restrict__ mod_w,
                           const float* __restrict__ mod_b,
                           float* __restrict__ s) {
  int idx = blockIdx.x * blockDim.x + threadIdx.x;     // b*512 + i
  if (idx >= B_ * CIN) return;
  int b = idx >> 9, i = idx & (CIN - 1);
  float acc = 0.f;
  for (int j = 0; j < 512; ++j)
    acc += style[b * 512 + j] * mod_w[i * 512 + j];
  s[idx] = acc * 0.04419417382415922f + mod_b[i];      // 1/sqrt(512)
}

// ---------------------------------------------------------------------------
// Prep kernel 2: q[o,i] = sum_k weight[o,i,k]^2
// ---------------------------------------------------------------------------
__global__ void q_kernel(const float* __restrict__ weight, float* __restrict__ q) {
  int idx = blockIdx.x * blockDim.x + threadIdx.x;     // o*512 + i
  if (idx >= COUT * CIN) return;
  float acc = 0.f;
#pragma unroll
  for (int k = 0; k < 9; ++k) {
    float w = weight[(size_t)idx * 9 + k];
    acc += w * w;
  }
  q[idx] = acc;
}

// ---------------------------------------------------------------------------
// Prep kernel 3: g[b,o] = scale * rsqrt(scale^2 * sum_i s^2 q + 1e-8)
// ---------------------------------------------------------------------------
__global__ void g_kernel(const float* __restrict__ s, const float* __restrict__ q,
                         float* __restrict__ g) {
  int idx = blockIdx.x * blockDim.x + threadIdx.x;     // b*256 + o
  if (idx >= B_ * COUT) return;
  int b = idx >> 8, o = idx & (COUT - 1);
  float acc = 0.f;
  for (int i = 0; i < CIN; ++i) {
    float sv = s[b * CIN + i];
    acc += sv * sv * q[o * CIN + i];
  }
  float demod = rsqrtf(acc * (1.0f / 4608.0f) + 1e-8f);
  g[idx] = demod * 0.014731391274719739f;              // 1/sqrt(4608)
}

// ---------------------------------------------------------------------------
// Prep kernel 4: xs, channel-last tiled layout for the TDM:
//   xs[b][kc][y][x][k32] = (f16)(s[b, kc*32+k] * x[b, kc*32+k, y, x])
// One (b,kc,y) row = 64*32 halves = 4KB contiguous  ->  TDM-friendly.
// ---------------------------------------------------------------------------
__global__ void xs_kernel(const float* __restrict__ x, const float* __restrict__ s,
                          half_t* __restrict__ xs) {
  size_t idx = (size_t)blockIdx.x * blockDim.x + threadIdx.x;
  if (idx >= (size_t)B_ * CIN * HW * HW) return;
  int k  = (int)(idx & 31);
  int xi = (int)((idx >> 5) & 63);
  int y  = (int)((idx >> 11) & 63);
  int kc = (int)((idx >> 17) & 15);
  int b  = (int)(idx >> 21);
  int i  = kc * 32 + k;
  float v = x[((size_t)(b * CIN + i) * HW + y) * HW + xi];
  xs[idx] = (half_t)(v * s[b * CIN + i]);
}

// ---------------------------------------------------------------------------
// Prep kernel 5: P0 = blur (4x4, sep [1,3,3,1]/4) convolved with weight (3x3),
// polyphase-split into 4 parities x 9 taps, packed f16 as
// [parity p][tap t][kchunk kc][o:256][kk:32]  (WMMA-A friendly)
// ---------------------------------------------------------------------------
__global__ void prepw_kernel(const float* __restrict__ weight, half_t* __restrict__ p0) {
  int idx = blockIdx.x * blockDim.x + threadIdx.x;     // o*512 + i
  if (idx >= COUT * CIN) return;
  int o = idx >> 9, i = idx & 511;
  float w[9];
#pragma unroll
  for (int k = 0; k < 9; ++k) w[k] = weight[(size_t)idx * 9 + k];
  const float k1[4] = {0.25f, 0.75f, 0.75f, 0.25f};
  int kc = i >> 5, kk = i & 31;
#pragma unroll
  for (int ty = 0; ty < 6; ++ty) {
#pragma unroll
    for (int tx = 0; tx < 6; ++tx) {
      float e = 0.f;
#pragma unroll
      for (int jy = 0; jy < 3; ++jy) {
        int cy = ty - jy;
        if (cy < 0 || cy > 3) continue;
#pragma unroll
        for (int jx = 0; jx < 3; ++jx) {
          int cx = tx - jx;
          if (cx < 0 || cx > 3) continue;
          e += k1[cy] * k1[cx] * w[jy * 3 + jx];
        }
      }
      int p = (ty & 1) * 2 + (tx & 1);
      int t = (ty >> 1) * 3 + (tx >> 1);
      p0[((size_t)((p * 9 + t) * KC_N + kc) * COUT + o) * 32 + kk] = (half_t)e;
    }
  }
}

// ---------------------------------------------------------------------------
// Main fused kernel. Block = (b, my, parity) x half of Cout; 8 waves.
// Per K-chunk: 2-3 tensor_load_to_lds row-DMAs (64 cols x 32 ch f16 each,
// contiguous in global, landing at column 1 of a 66-col padded LDS row),
// double-buffered and overlapped with 36 WMMAs. Pad columns / halo rows are
// zeroed once, so the inner loop has no edge-case VALU at all.
// ---------------------------------------------------------------------------
__global__ __launch_bounds__(256)
void synth_wmma(const half_t* __restrict__ xs,
                const half_t* __restrict__ p0,
                const float*  __restrict__ g,
                const float*  __restrict__ noise,
                const float*  __restrict__ noise_w,
                const float*  __restrict__ act_b,
                float* __restrict__ out) {
  __shared__ half_t lds[2][BUF_H];                     // 2 x 12.4 KB ping-pong

  const int tid  = threadIdx.x;
  const int lane = tid & 31;
  const int wave = tid >> 5;
  const int lh   = lane & 15;
  const bool hi  = lane >= 16;

  const int bid = blockIdx.x;
  const int pp  = bid & 3;                             // parity (py,px)
  const int my  = (bid >> 2) & 63;                     // output row / 2
  const int b   = bid >> 8;
  const int obase = blockIdx.y * 128 + wave * 16;

  v8f acc0 = {}, acc1 = {}, acc2 = {}, acc3 = {};

  const int a_kk0 = hi ? 8 : 0;                        // A: lanes16-31 -> K+8 / K+24
  const int b_k0  = hi ? 8 : 0;                        // B: lanes16-31 -> K+8 / K+24

  // generic pointer low 32 bits == wave-relative LDS byte offset
  const uint32_t lds_off = (uint32_t)(uintptr_t)(&lds[0][0]);

  const int r0 = (my == 0) ? 1 : 0;                    // valid halo rows [r0, r1]
  const int r1 = (my == 63) ? 1 : 2;

  // one-time zeroing: pad columns c=0 and c=65 (both buffers, all rows)
  for (int e = tid; e < 2 * 3 * 2 * 32; e += 256) {    // 384 halves
    int kk = e & 31;
    int cc = (e >> 5) & 1;                             // 0 -> col 0, 1 -> col 65
    int rr = (e >> 6) % 3;
    int bb = e / 192;
    lds[bb][(rr * 66 + cc * 65) * 32 + kk] = (half_t)0.f;
  }
  // one-time zeroing: out-of-range halo rows (never touched by the DMA)
  if (my == 0)
    for (int e = tid; e < ROW_H; e += 256) { lds[0][e] = (half_t)0.f; lds[1][e] = (half_t)0.f; }
  if (my == 63)
    for (int e = tid; e < ROW_H; e += 256) { lds[0][2 * ROW_H + e] = (half_t)0.f; lds[1][2 * ROW_H + e] = (half_t)0.f; }

  // ---- Tensor Data Mover: one 1-D row descriptor per valid input row ----
  auto tdm_issue = [&](int kc, int buf) {
    for (int r = r0; r <= r1; ++r) {
      uint64_t ga = (uint64_t)(uintptr_t)(xs + ((size_t)((b * 16 + kc) * 64 + my - 1 + r) * 2048));
      u32x4 g0;
      g0[0] = 1u;                                      // count=1, user load
      g0[1] = lds_off + (uint32_t)buf * BUF_B + (uint32_t)r * ROW_B + 64u; // col 1
      g0[2] = (uint32_t)ga;                            // global_addr[31:0]
      g0[3] = ((uint32_t)(ga >> 32) & 0x01FFFFFFu) | 0x80000000u; // addr[56:32] | type=2
      i32x8 g1;
      g1[0] = (int)(1u << 16);      // workgroup_mask=0, data_size=1 (2 bytes)
      g1[1] = (int)(2048u << 16);   // tensor_dim0[15:0] = 2048
      g1[2] = (int)(1u << 16);      // tensor_dim0[31:16]=0, tensor_dim1[15:0]=1
      g1[3] = (int)(2048u << 16);   // tensor_dim1[31:16]=0, tile_dim0 = 2048
      g1[4] = 1;                    // tile_dim1 = 1, tile_dim2 = 0
      g1[5] = 2048;                 // tensor_dim0_stride[31:0]
      g1[6] = 0;
      g1[7] = 0;
      i32x4 gz4 = {0, 0, 0, 0};
#if __clang_major__ >= 23
      i32x8 gz8 = {0, 0, 0, 0, 0, 0, 0, 0};
      __builtin_amdgcn_tensor_load_to_lds(g0, g1, gz4, gz4, gz8, 0);
#else
      __builtin_amdgcn_tensor_load_to_lds(g0, g1, gz4, gz4, 0);
#endif
    }
  };

  if (wave == 0) tdm_issue(0, 0);                      // prime the pipeline

  for (int kc = 0; kc < KC_N; ++kc) {
    if (wave == 0) __builtin_amdgcn_s_wait_tensorcnt(0);
    __syncthreads();                                   // publish DMA'd tile
    if (wave == 0 && kc + 1 < KC_N)
      tdm_issue(kc + 1, (kc + 1) & 1);                 // overlap next DMA

    const half_t* buf = &lds[kc & 1][0];
    if (kc + 1 < KC_N)                                 // prefetch next A block
      __builtin_prefetch(p0 + ((size_t)((pp * 9 * KC_N) + kc + 1) * COUT + obase + lane) * 32, 0, 3);

#pragma unroll
    for (int t = 0; t < 9; ++t) {
      const int dy = t / 3, dx = t % 3;
      // ---- A fragment: P0[pp][t][kc][obase+lh][a_kk0..+7, +16..+23] ----
      const half_t* ap =
          p0 + ((size_t)(((pp * 9 + t) * KC_N + kc) * COUT) + obase + lh) * 32 + a_kk0;
      v8h alo = *(const v8h*)(ap);
      v8h ahi = *(const v8h*)(ap + 16);
      v16h afrag = __builtin_shufflevector(alo, ahi,
          0, 1, 2, 3, 4, 5, 6, 7, 8, 9, 10, 11, 12, 13, 14, 15);

      const int ry = 2 - dy;                           // input row = my + 1 - dy
#pragma unroll
      for (int nt = 0; nt < 4; ++nt) {
        const int n = nt * 16 + lh;                    // output col index mx
        // padded LDS column = (input col) + 1 = n + 2 - dx, always in [0, 65]
        const half_t* bp = buf + ((ry * 66 + (n + 2 - dx)) * 32 + b_k0);
        v8h blo = *(const v8h*)(bp);
        v8h bhi = *(const v8h*)(bp + 16);
        v16h bfrag = __builtin_shufflevector(blo, bhi,
            0, 1, 2, 3, 4, 5, 6, 7, 8, 9, 10, 11, 12, 13, 14, 15);
        v8f& acc = (nt == 0 ? acc0 : nt == 1 ? acc1 : nt == 2 ? acc2 : acc3);
        acc = __builtin_amdgcn_wmma_f32_16x16x32_f16(
            false, afrag, false, bfrag, (short)0, acc, false, false);
      }
    }
    // no trailing barrier: next DMA targets the other buffer, and the top-of-
    // loop barrier orders all readers of that buffer before it is refilled
  }

  // ---- epilogue: scale by g, add noise + bias, leaky-ReLU * sqrt(2) ----
  const int py = pp >> 1, px = pp & 1;
  const int y  = 2 * my + py;
  const float nw = noise_w[0];
  auto epi = [&](const v8f& acc, int nt) {
    int n  = nt * 16 + lh;
    int xo = 2 * n + px;
    float nz = nw * noise[((size_t)b * OHW + y) * OHW + xo];
#pragma unroll
    for (int r = 0; r < 8; ++r) {
      int o = obase + (hi ? 8 : 0) + r;                // C layout: M = r (+8 hi half)
      float val = acc[r] * g[b * COUT + o] + nz + act_b[o];
      val = (val > 0.f ? val : 0.2f * val) * 1.4142135623730951f;
      out[(((size_t)b * COUT + o) * OHW + y) * OHW + xo] = val;
    }
  };
  epi(acc0, 0); epi(acc1, 1); epi(acc2, 2); epi(acc3, 3);
}

// ---------------------------------------------------------------------------
// Launch
// ---------------------------------------------------------------------------
extern "C" void kernel_launch(void* const* d_in, const int* in_sizes, int n_in,
                              void* d_out, int out_size, void* d_ws, size_t ws_size,
                              hipStream_t stream) {
  const float* x       = (const float*)d_in[0];   // [8,512,64,64]
  const float* style   = (const float*)d_in[1];   // [8,512]
  const float* noise   = (const float*)d_in[2];   // [8,1,128,128]
  const float* weight  = (const float*)d_in[3];   // [1,256,512,3,3]
  const float* mod_w   = (const float*)d_in[4];   // [512,512]
  const float* mod_b   = (const float*)d_in[5];   // [512]
  const float* noise_w = (const float*)d_in[6];   // [1]
  const float* act_b   = (const float*)d_in[7];   // [1,256,1,1]
  float* out = (float*)d_out;                     // [8,256,128,128]

  char* ws = (char*)d_ws;
  float*  s_buf = (float*)(ws + 0);               //  16 KB
  float*  q_buf = (float*)(ws + 16384);           // 512 KB
  float*  g_buf = (float*)(ws + 540672);          //   8 KB
  half_t* xs    = (half_t*)(ws + 548864);         //  32 MB (f16, TDM-tiled)
  half_t* p0    = (half_t*)(ws + 34103296);       //   9 MB (f16)  total ~43.5 MB

  mod_kernel  <<<(B_ * CIN + 255) / 256, 256, 0, stream>>>(style, mod_w, mod_b, s_buf);
  q_kernel    <<<(COUT * CIN) / 256,     256, 0, stream>>>(weight, q_buf);
  g_kernel    <<<(B_ * COUT) / 256,      256, 0, stream>>>(s_buf, q_buf, g_buf);
  xs_kernel   <<<(B_ * CIN * HW * HW) / 256, 256, 0, stream>>>(x, s_buf, xs);
  prepw_kernel<<<(COUT * CIN) / 256,     256, 0, stream>>>(weight, p0);

  dim3 grid(B_ * 64 * 4, 2);                      // (b, my, parity) x Cout halves
  synth_wmma<<<grid, 256, 0, stream>>>(xs, p0, g_buf, noise, noise_w, act_b, out);
}
